// ClassificationMPNSimple_65979287601856
// MI455X (gfx1250) — compile-verified, matrix operand order checked by reading, not run on hardware
//
#include <hip/hip_runtime.h>

typedef __attribute__((ext_vector_type(16))) _Float16 v16h;
typedef __attribute__((ext_vector_type(8)))  _Float16 v8h;
typedef __attribute__((ext_vector_type(8)))  float    v8f;
typedef __attribute__((ext_vector_type(4)))  float    v4f;

#define DEVINL __device__ __forceinline__

DEVINL v16h cat8(v8h lo, v8h hi) {
  return __builtin_shufflevector(lo, hi, 0,1,2,3,4,5,6,7,8,9,10,11,12,13,14,15);
}

// CDNA5 16-bit A/B operand fragment: lane r = row (M for A, N for B-transposed),
// half-select picks K chunks {hs*8 .. +7} and {16+hs*8 .. +7} within a 32-wide k-tile.
DEVINL v16h load_frag(const _Float16* mat, int ld, int kt) {
  int lane = threadIdx.x & 31;
  int r  = lane & 15;
  int hs = lane >> 4;
  const _Float16* p = mat + (size_t)r * ld + kt * 32 + hs * 8;
  v8h lo = *(const v8h*)(p);
  v8h hi = *(const v8h*)(p + 16);
  return cat8(lo, hi);
}

DEVINL v8f wmma_step(v16h a, v16h b, v8f c) {
  return __builtin_amdgcn_wmma_f32_16x16x32_f16(false, a, false, b, (short)0, c, false, false);
}

// C/D layout: lane half 0 -> M = g, half 1 -> M = g + 8; N = (lane & 15).
DEVINL void store_tile_lds_relu(v8f acc, _Float16* out, int ldo, int nt, const float* bias) {
  int lane  = threadIdx.x & 31;
  int n     = (nt << 4) | (lane & 15);
  int mbase = (lane >> 4) << 3;
  float bv = bias[n];
#pragma unroll
  for (int g = 0; g < 8; ++g) {
    float v = fmaxf(acc[g] + bv, 0.0f);
    out[(mbase + g) * ldo + n] = (_Float16)v;
  }
}

// ---- weight convert: W[K][N] fp32 -> Wt[N][K] f16 (transposed, WMMA-B friendly) ----
__global__ void wcvt_kernel(const float* __restrict__ W, _Float16* __restrict__ Wt, int K, int N) {
  int total = K * N;
  for (int idx = blockIdx.x * blockDim.x + threadIdx.x; idx < total; idx += gridDim.x * blockDim.x) {
    int n = idx / K, k = idx % K;
    Wt[idx] = (_Float16)W[(size_t)k * N + n];
  }
}

__global__ void zero_kernel(float* __restrict__ p, int n) {
  for (int i = blockIdx.x * blockDim.x + threadIdx.x; i < n; i += gridDim.x * blockDim.x) p[i] = 0.0f;
}

// ---- fused 2-layer MLP embed: out = relu(relu(in@W1+b1)@W2+b2), in fp32 [rows,Kin] ----
__global__ __launch_bounds__(128) void embed2_kernel(
    const float* __restrict__ in, int Kin,
    const _Float16* __restrict__ W1t, const float* __restrict__ b1,
    const _Float16* __restrict__ W2t, const float* __restrict__ b2,
    _Float16* __restrict__ out0, _Float16* __restrict__ out1, int Nrows) {
  __shared__ _Float16 At[64 * 128];
  __shared__ _Float16 Ht[64 * 128];
  int tid = threadIdx.x;
  int row0 = blockIdx.x * 64;
  int cpr = Kin >> 2;                       // float4 chunks per row
  for (int idx = tid; idx < 64 * cpr; idx += 128) {
    int r = idx / cpr, c4 = (idx % cpr) << 2;
    int row = row0 + r;
    v4f v = {};
    if (row < Nrows) v = *(const v4f*)(in + (size_t)row * Kin + c4);
    _Float16* d = &At[r * Kin + c4];
    d[0] = (_Float16)v[0]; d[1] = (_Float16)v[1];
    d[2] = (_Float16)v[2]; d[3] = (_Float16)v[3];
  }
  __syncthreads();
  int w = tid >> 5, lane = tid & 31;
  const _Float16* Asub = &At[(w * 16) * Kin];
  _Float16* Hsub = &Ht[(w * 16) * 128];
  int kt1 = Kin >> 5;
  for (int nt = 0; nt < 8; ++nt) {
    v8f acc = {};
    for (int kt = 0; kt < kt1; ++kt)
      acc = wmma_step(load_frag(Asub, Kin, kt),
                      load_frag(W1t + (size_t)(nt * 16) * Kin, Kin, kt), acc);
    store_tile_lds_relu(acc, Hsub, 128, nt, b1);
  }
  __syncthreads();
  for (int nt = 0; nt < 8; ++nt) {
    v8f acc = {};
    for (int kt = 0; kt < 4; ++kt)
      acc = wmma_step(load_frag(Hsub, 128, kt),
                      load_frag(W2t + (size_t)(nt * 16) * 128, 128, kt), acc);
    int n = (nt << 4) | (lane & 15);
    int mbase = (lane >> 4) << 3;
    float bv = b2[n];
#pragma unroll
    for (int g = 0; g < 8; ++g) {
      int row = row0 + w * 16 + mbase + g;
      if (row < Nrows) {
        _Float16 h = (_Float16)fmaxf(acc[g] + bv, 0.0f);
        out0[(size_t)row * 128 + n] = h;
        out1[(size_t)row * 128 + n] = h;
      }
    }
  }
}

// ---- fused edge step: build A=[x_i|x_j|efc] (16x768), e=relu(relu(A@We1)@We2),
//      m=relu([x_i|e]@Wn1), agg[i] += m, ef <- e (in place) ----
__global__ __launch_bounds__(128) void edge_step_kernel(
    const _Float16* __restrict__ nf0, const _Float16* __restrict__ nf,
    const _Float16* __restrict__ ef0, _Float16* __restrict__ ef,
    const int* __restrict__ jsrc, const int* __restrict__ itgt,
    const _Float16* __restrict__ We1t, const float* __restrict__ be1,
    const _Float16* __restrict__ We2t, const float* __restrict__ be2,
    const _Float16* __restrict__ Wn1t, const float* __restrict__ bn1,
    float* __restrict__ agg) {
  __shared__ _Float16 A[16 * 768];
  __shared__ _Float16 H1[16 * 512];
  __shared__ _Float16 Et[16 * 128];
  __shared__ int si[16], sj[16];
  int tid = threadIdx.x;
  int e0 = blockIdx.x * 16;
  if (tid < 16) { si[tid] = itgt[e0 + tid]; sj[tid] = jsrc[e0 + tid]; }
  __syncthreads();
  for (int c = tid; c < 16 * 96; c += 128) {
    int r = c / 96;
    int col = (c % 96) << 3;
    const _Float16* src;
    if      (col < 128) src = nf0 + (size_t)si[r] * 128 + col;
    else if (col < 256) src = nf  + (size_t)si[r] * 128 + (col - 128);
    else if (col < 384) src = nf0 + (size_t)sj[r] * 128 + (col - 256);
    else if (col < 512) src = nf  + (size_t)sj[r] * 128 + (col - 384);
    else if (col < 640) src = ef0 + (size_t)(e0 + r) * 128 + (col - 512);
    else                src = ef  + (size_t)(e0 + r) * 128 + (col - 640);
    *(v8h*)&A[r * 768 + col] = *(const v8h*)src;
  }
  __syncthreads();
  int w = tid >> 5, lane = tid & 31;
  // GEMM1: 16x768 @ 768x512
  for (int t = 0; t < 8; ++t) {
    int nt = w + t * 4;
    v8f acc = {};
    for (int kt = 0; kt < 24; ++kt)
      acc = wmma_step(load_frag(A, 768, kt),
                      load_frag(We1t + (size_t)(nt * 16) * 768, 768, kt), acc);
    store_tile_lds_relu(acc, H1, 512, nt, be1);
  }
  __syncthreads();
  // GEMM2: 16x512 @ 512x128 -> e (LDS + global ef in place)
  for (int t = 0; t < 2; ++t) {
    int nt = w + t * 4;
    v8f acc = {};
    for (int kt = 0; kt < 16; ++kt)
      acc = wmma_step(load_frag(H1, 512, kt),
                      load_frag(We2t + (size_t)(nt * 16) * 512, 512, kt), acc);
    int n = (nt << 4) | (lane & 15);
    int mbase = (lane >> 4) << 3;
    float bv = be2[n];
#pragma unroll
    for (int g = 0; g < 8; ++g) {
      _Float16 h = (_Float16)fmaxf(acc[g] + bv, 0.0f);
      Et[(mbase + g) * 128 + n] = h;
      ef[(size_t)(e0 + mbase + g) * 128 + n] = h;
    }
  }
  __syncthreads();
  // GEMM3: m = relu([x_i (A cols 0..255) | e (Et)] @ Wn1), scatter-add to agg[i]
  for (int t = 0; t < 2; ++t) {
    int nt = w + t * 4;
    v8f acc = {};
    for (int kt = 0; kt < 12; ++kt) {
      v16h a = (kt < 8) ? load_frag(A, 768, kt) : load_frag(Et, 128, kt - 8);
      acc = wmma_step(a, load_frag(Wn1t + (size_t)(nt * 16) * 384, 384, kt), acc);
    }
    int n = (nt << 4) | (lane & 15);
    int mbase = (lane >> 4) << 3;
    float bv = bn1[n];
#pragma unroll
    for (int g = 0; g < 8; ++g) {
      float v = fmaxf(acc[g] + bv, 0.0f);
      int node = si[mbase + g];
      __hip_atomic_fetch_add(&agg[(size_t)node * 128 + n], v,
                             __ATOMIC_RELAXED, __HIP_MEMORY_SCOPE_AGENT);
    }
  }
}

// ---- node update: nf = relu(agg @ Wu + bu), agg fp32 ----
__global__ __launch_bounds__(128) void update_kernel(
    const float* __restrict__ in, const _Float16* __restrict__ Wt,
    const float* __restrict__ b, _Float16* __restrict__ out, int Nrows) {
  __shared__ _Float16 At[64 * 128];
  int tid = threadIdx.x;
  int row0 = blockIdx.x * 64;
  for (int idx = tid; idx < 64 * 32; idx += 128) {
    int r = idx >> 5, c4 = (idx & 31) << 2;
    int row = row0 + r;
    v4f v = {};
    if (row < Nrows) v = *(const v4f*)(in + (size_t)row * 128 + c4);
    _Float16* d = &At[r * 128 + c4];
    d[0] = (_Float16)v[0]; d[1] = (_Float16)v[1];
    d[2] = (_Float16)v[2]; d[3] = (_Float16)v[3];
  }
  __syncthreads();
  int w = tid >> 5, lane = tid & 31;
  const _Float16* Asub = &At[(w * 16) * 128];
  for (int nt = 0; nt < 8; ++nt) {
    v8f acc = {};
    for (int kt = 0; kt < 4; ++kt)
      acc = wmma_step(load_frag(Asub, 128, kt),
                      load_frag(Wt + (size_t)(nt * 16) * 128, 128, kt), acc);
    int n = (nt << 4) | (lane & 15);
    int mbase = (lane >> 4) << 3;
    float bv = b[n];
#pragma unroll
    for (int g = 0; g < 8; ++g) {
      int row = row0 + w * 16 + mbase + g;
      if (row < Nrows) out[(size_t)row * 128 + n] = (_Float16)fmaxf(acc[g] + bv, 0.0f);
    }
  }
}

// ---- classification head: out = relu(feat@W1+b1) @ W2 + b2 (W2 is 128x1, fp32) ----
__global__ __launch_bounds__(128) void cls_kernel(
    const _Float16* __restrict__ feat,
    const _Float16* __restrict__ W1t, const float* __restrict__ b1,
    const float* __restrict__ W2, const float* __restrict__ b2,
    float* __restrict__ out, int Nrows) {
  __shared__ _Float16 At[64 * 128];
  __shared__ _Float16 Ht[64 * 128];
  __shared__ float partial[128];
  int tid = threadIdx.x;
  int row0 = blockIdx.x * 64;
  for (int idx = tid; idx < 64 * 16; idx += 128) {
    int r = idx >> 4, c8 = (idx & 15) << 3;
    int row = row0 + r;
    v8h v = {};
    if (row < Nrows) v = *(const v8h*)(feat + (size_t)row * 128 + c8);
    *(v8h*)&At[r * 128 + c8] = v;
  }
  __syncthreads();
  int w = tid >> 5;
  const _Float16* Asub = &At[(w * 16) * 128];
  _Float16* Hsub = &Ht[(w * 16) * 128];
  for (int nt = 0; nt < 8; ++nt) {
    v8f acc = {};
    for (int kt = 0; kt < 4; ++kt)
      acc = wmma_step(load_frag(Asub, 128, kt),
                      load_frag(W1t + (size_t)(nt * 16) * 128, 128, kt), acc);
    store_tile_lds_relu(acc, Hsub, 128, nt, b1);
  }
  __syncthreads();
  int r = tid >> 1, hs = tid & 1;
  float s = 0.0f;
  for (int c = 0; c < 64; ++c) {
    int col = hs * 64 + c;
    s += (float)Ht[r * 128 + col] * W2[col];
  }
  partial[tid] = s;
  __syncthreads();
  if (hs == 0) {
    int row = row0 + r;
    if (row < Nrows) out[row] = partial[tid] + partial[tid + 1] + b2[0];
  }
}

extern "C" void kernel_launch(void* const* d_in, const int* in_sizes, int n_in,
                              void* d_out, int out_size, void* d_ws, size_t ws_size,
                              hipStream_t stream) {
  (void)in_sizes; (void)n_in; (void)out_size; (void)ws_size;
  const int N = 10000, E = 200000;
  const float* x         = (const float*)d_in[0];
  const float* edge_attr = (const float*)d_in[1];
  const int*   eidx      = (const int*)d_in[2];
  const float* Wne1 = (const float*)d_in[3];  const float* bne1 = (const float*)d_in[4];
  const float* Wne2 = (const float*)d_in[5];  const float* bne2 = (const float*)d_in[6];
  const float* Wee1 = (const float*)d_in[7];  const float* bee1 = (const float*)d_in[8];
  const float* Wee2 = (const float*)d_in[9];  const float* bee2 = (const float*)d_in[10];
  const float* We1  = (const float*)d_in[11]; const float* be1  = (const float*)d_in[12];
  const float* We2  = (const float*)d_in[13]; const float* be2  = (const float*)d_in[14];
  const float* Wn1  = (const float*)d_in[15]; const float* bn1  = (const float*)d_in[16];
  const float* Wu   = (const float*)d_in[17]; const float* bu   = (const float*)d_in[18];
  const float* Wnc1 = (const float*)d_in[19]; const float* bnc1 = (const float*)d_in[20];
  const float* Wnc2 = (const float*)d_in[21]; const float* bnc2 = (const float*)d_in[22];
  const float* Wec1 = (const float*)d_in[23]; const float* bec1 = (const float*)d_in[24];
  const float* Wec2 = (const float*)d_in[25]; const float* bec2 = (const float*)d_in[26];
  const int* jsrc = eidx;      // edge_index[0] = source
  const int* itgt = eidx + E;  // edge_index[1] = target

  char* ws = (char*)d_ws;
  size_t off = 0;
  auto take = [&](size_t bytes) -> void* {
    void* p = ws + off;
    off = (off + bytes + 255) & ~(size_t)255;
    return p;
  };
  _Float16* nf0 = (_Float16*)take((size_t)N * 128 * 2);
  _Float16* nf  = (_Float16*)take((size_t)N * 128 * 2);
  _Float16* ef0 = (_Float16*)take((size_t)E * 128 * 2);
  _Float16* ef  = (_Float16*)take((size_t)E * 128 * 2);
  float*    agg = (float*)take((size_t)N * 128 * 4);
  _Float16* Wne1t = (_Float16*)take(128 * 128 * 2);
  _Float16* Wne2t = (_Float16*)take(128 * 128 * 2);
  _Float16* Wee1t = (_Float16*)take(128 * 64 * 2);
  _Float16* Wee2t = (_Float16*)take(128 * 128 * 2);
  _Float16* We1t  = (_Float16*)take((size_t)512 * 768 * 2);
  _Float16* We2t  = (_Float16*)take((size_t)128 * 512 * 2);
  _Float16* Wn1t  = (_Float16*)take((size_t)128 * 384 * 2);
  _Float16* Wut   = (_Float16*)take(128 * 128 * 2);
  _Float16* Wnc1t = (_Float16*)take(128 * 128 * 2);
  _Float16* Wec1t = (_Float16*)take(128 * 128 * 2);

  auto cvt = [&](const float* W, _Float16* Wt, int K, int Ncols) {
    int total = K * Ncols;
    wcvt_kernel<<<(total + 255) / 256, 256, 0, stream>>>(W, Wt, K, Ncols);
  };
  cvt(Wne1, Wne1t, 128, 128);
  cvt(Wne2, Wne2t, 128, 128);
  cvt(Wee1, Wee1t, 64, 128);
  cvt(Wee2, Wee2t, 128, 128);
  cvt(We1,  We1t,  768, 512);
  cvt(We2,  We2t,  512, 128);
  cvt(Wn1,  Wn1t,  384, 128);
  cvt(Wu,   Wut,   128, 128);
  cvt(Wnc1, Wnc1t, 128, 128);
  cvt(Wec1, Wec1t, 128, 128);

  embed2_kernel<<<(N + 63) / 64, 128, 0, stream>>>(x, 128, Wne1t, bne1, Wne2t, bne2, nf0, nf, N);
  embed2_kernel<<<(E + 63) / 64, 128, 0, stream>>>(edge_attr, 64, Wee1t, bee1, Wee2t, bee2, ef0, ef, E);

  for (int s = 0; s < 2; ++s) {
    zero_kernel<<<(N * 128 + 255) / 256, 256, 0, stream>>>(agg, N * 128);
    edge_step_kernel<<<E / 16, 128, 0, stream>>>(nf0, nf, ef0, ef, jsrc, itgt,
                                                 We1t, be1, We2t, be2, Wn1t, bn1, agg);
    update_kernel<<<(N + 63) / 64, 128, 0, stream>>>(agg, Wut, bu, nf, N);
  }

  float* out_edge = (float*)d_out;
  float* out_node = (float*)d_out + E;
  cls_kernel<<<(E + 63) / 64, 128, 0, stream>>>(ef, Wec1t, bec1, Wec2, bec2, out_edge, E);
  cls_kernel<<<(N + 63) / 64, 128, 0, stream>>>(nf, Wnc1t, bnc1, Wnc2, bnc2, out_node, N);
}
